// RSSM_21277267985046
// MI455X (gfx1250) — compile-verified, matrix operand order checked by reading, not run on hardware
//
#include <hip/hip_runtime.h>
#include <hip/hip_bf16.h>

// Flip to 0 if the assembler rejects the async mnemonics.
#define USE_ASYNC 1

// ---------------- types ----------------
typedef __bf16 v16bf __attribute__((ext_vector_type(16)));
typedef float  v8f   __attribute__((ext_vector_type(8)));
typedef unsigned int u32x4 __attribute__((ext_vector_type(4)));

#define DEV __device__ __forceinline__

// Problem dims
#define BB   256
#define TT   64
#define EMB  4096
#define ACTD 6
#define HH   512
#define S2   1024
#define G3   1536
#define KREC 1056   /* 1024 (z) + 6 (act), padded to mult of 32 */
#define KREP 4608   /* 4096 (symlog embed) + 512 (h) */

// Epilogue modes
#define EPI_BIAS_F32    0
#define EPI_BIAS_BF16   1
#define EPI_LNSILU_BF16 2
#define EPI_LN_F32      3

static DEV v8f wmma_bf16(v16bf a, v16bf b, v8f c) {
  // D = A(16x32 bf16) x B(32x16 bf16) + C(16x16 f32)
  return __builtin_amdgcn_wmma_f32_16x16x32_bf16(false, a, false, b,
                                                 (short)0, c, false, false);
}

// assemble v16bf operand from two 16-byte chunks (per-lane K runs of 8)
static DEV v16bf mk16(const __hip_bfloat16* p0, const __hip_bfloat16* p1) {
  union { v16bf v; u32x4 q[2]; } u;
  u.q[0] = *(const u32x4*)p0;
  u.q[1] = *(const u32x4*)p1;
  return u.v;
}

// 16-byte global -> LDS copy. Async path uses the CDNA5 Async DMA
// (GLOBAL_LOAD_ASYNC_TO_LDS_B128, tracked by ASYNCcnt; LDS byte address in
// vdst VGPR = low 32 bits of the generic shared pointer per the aperture rule).
static DEV void copy16_g2l(__hip_bfloat16* lds_dst, const __hip_bfloat16* gsrc) {
#if USE_ASYNC
  const unsigned lds = (unsigned)(size_t)lds_dst;
  const unsigned long long ga = (unsigned long long)(size_t)gsrc;
  asm volatile("global_load_async_to_lds_b128 %0, %1, off"
               :: "v"(lds), "v"(ga) : "memory");
#else
  const u32x4 v = *(const u32x4*)gsrc;
  *(u32x4*)lds_dst = v;
#endif
}

static DEV void g2l_wait() {
#if USE_ASYNC
  asm volatile("s_wait_asynccnt 0x0" ::: "memory");
#endif
}

// ---------------- fused WMMA GEMM: out = epi(A[M,K] @ Wt[N,K]^T + b) ----------
// grid = M/16 blocks of 256 threads (8 waves). Each wave owns NTILES/8 N-tiles,
// so a block covers the whole N row => LayerNorm can reduce in-block.
template<int EPI, int NTILES>
__global__ __launch_bounds__(256)
void gemm16(const __hip_bfloat16* __restrict__ A, int lda,       // lda == K (padded)
            const __hip_bfloat16* __restrict__ Wt,               // [N][K] bf16
            const float* __restrict__ bias,
            const float* __restrict__ gamma,
            const float* __restrict__ beta,
            float* __restrict__ outF,
            __hip_bfloat16* __restrict__ outB,
            int ostride, int col0)
{
  constexpr int WPB = 8;
  constexpr int NT  = NTILES / WPB;
  constexpr int N   = NTILES * 16;

  const int tid  = threadIdx.x;
  const int wid  = tid >> 5;
  const int lane = tid & 31;
  const int h    = lane >> 4;     // lane half selects K sub-run / M+8 rows
  const int l16  = lane & 15;
  const int m0   = blockIdx.x * 16;
  const int K    = lda;
  const int nkt  = K / 32;

  __shared__ __align__(16) __hip_bfloat16 sA[2][16 * 32];
  __shared__ float rsum[16], rsq[16];

  if constexpr (EPI == EPI_LNSILU_BF16 || EPI == EPI_LN_F32) {
    if (tid < 16) { rsum[tid] = 0.0f; rsq[tid] = 0.0f; }
  }

  v8f acc[NT];
  const v8f vzero = {0.f,0.f,0.f,0.f,0.f,0.f,0.f,0.f};
#pragma unroll
  for (int t = 0; t < NT; ++t) acc[t] = vzero;

  // cooperative A-tile stage (16 rows x 32 bf16 = 1KB), 64 lanes x 16B
  auto stage = [&](int buf, int kt) {
    if (tid < 64) {
      const int r = tid >> 2, s = tid & 3;
      copy16_g2l(&sA[buf][r * 32 + s * 8],
                 A + (long)(m0 + r) * lda + kt * 32 + s * 8);
    }
  };

  stage(0, 0);
  g2l_wait();
  __syncthreads();

  for (int kt = 0; kt < nkt; ++kt) {
    const int cur = kt & 1;
    // kick off async copy of the next A tile; overlaps with WMMA below
    if (kt + 1 < nkt) stage(cur ^ 1, kt + 1);
    // prefetch next-K weight block into GL2 (global_prefetch_b8)
    if (kt + 1 < nkt && lane == 0)
      __builtin_prefetch(Wt + (long)(wid * 16) * K + (kt + 1) * 32, 0, 1);

    const __hip_bfloat16* arow = &sA[cur][l16 * 32 + 8 * h];
    const v16bf av = mk16(arow, arow + 16);

#pragma unroll
    for (int t = 0; t < NT; ++t) {
      const int nt = wid + t * WPB;
      const __hip_bfloat16* bp = Wt + (long)(nt * 16 + l16) * K + kt * 32 + 8 * h;
      const v16bf bv = mk16(bp, bp + 16);
      acc[t] = wmma_bf16(av, bv, acc[t]);
    }

    g2l_wait();          // next tile landed in LDS
    __syncthreads();     // visible to all waves; prev buffer free to overwrite
  }

  // ---- epilogue: bias (pre-LN, per reference) ----
#pragma unroll
  for (int t = 0; t < NT; ++t) {
    const int n = (wid + t * WPB) * 16 + l16;
    const float bz = bias[n];
#pragma unroll
    for (int r = 0; r < 8; ++r) acc[t][r] += bz;
  }

  if constexpr (EPI == EPI_LNSILU_BF16 || EPI == EPI_LN_F32) {
    // per-row mean/var over full N: wave shfl partials + LDS atomics
#pragma unroll
    for (int r = 0; r < 8; ++r) {
      float s = 0.f, q = 0.f;
#pragma unroll
      for (int t = 0; t < NT; ++t) { const float v = acc[t][r]; s += v; q += v * v; }
      for (int off = 8; off; off >>= 1) {
        s += __shfl_xor(s, off, 32);
        q += __shfl_xor(q, off, 32);
      }
      if (l16 == 0) { atomicAdd(&rsum[r + 8 * h], s); atomicAdd(&rsq[r + 8 * h], q); }
    }
    __syncthreads();

    float mu_[8], rs_[8];
    const float invN = 1.0f / (float)N;
#pragma unroll
    for (int r = 0; r < 8; ++r) {
      const float mu  = rsum[r + 8 * h] * invN;
      const float var = rsq[r + 8 * h] * invN - mu * mu;
      mu_[r] = mu;
      rs_[r] = rsqrtf(var + 1e-5f);
    }

#pragma unroll
    for (int t = 0; t < NT; ++t) {
      const int n = (wid + t * WPB) * 16 + l16;
      const float gam = gamma[n], bet = beta[n];
#pragma unroll
      for (int r = 0; r < 8; ++r) {
        const int m = m0 + r + 8 * h;
        float v = (acc[t][r] - mu_[r]) * rs_[r] * gam + bet;
        if constexpr (EPI == EPI_LNSILU_BF16) {
          v = v * (1.0f / (1.0f + __expf(-v)));          // SiLU
          outB[(long)m * ostride + col0 + n] = __float2bfloat16(v);
        } else {
          outF[(long)m * ostride + n] = v;
        }
      }
    }
  } else {
#pragma unroll
    for (int t = 0; t < NT; ++t) {
      const int n = (wid + t * WPB) * 16 + l16;
#pragma unroll
      for (int r = 0; r < 8; ++r) {
        const int m = m0 + r + 8 * h;
        const float v = acc[t][r];
        if constexpr (EPI == EPI_BIAS_BF16)
          outB[(long)m * ostride + col0 + n] = __float2bfloat16(v);
        else
          outF[(long)m * ostride + n] = v;
      }
    }
  }
}

// ---------------- weight convert: f32 W[K,N] -> bf16 Wt[N][KP] (zero-padded K)
__global__ void k_wt(const float* __restrict__ W, __hip_bfloat16* __restrict__ Wt,
                     int K, int N, int KP)
{
  const long i = (long)blockIdx.x * 256 + threadIdx.x;
  if (i < (long)N * KP) {
    const int n = (int)(i / KP), k = (int)(i % KP);
    const float v = (k < K) ? W[(long)k * N + n] : 0.0f;
    Wt[i] = __float2bfloat16(v);
  }
}

// ---------------- init: h0/z0 -> state buffers ----------------
__global__ void k_init(const float* __restrict__ h0, const float* __restrict__ z0,
                       float* __restrict__ hF, __hip_bfloat16* __restrict__ hB,
                       __hip_bfloat16* __restrict__ gin,
                       __hip_bfloat16* __restrict__ zin,
                       __hip_bfloat16* __restrict__ repin)
{
  const int i = blockIdx.x * 256 + threadIdx.x;
  if (i < BB * KREC) {
    const int b = i / KREC, c = i % KREC;
    const float v = (c < S2) ? z0[b * S2 + c] : 0.0f;  // pad cols stay 0
    zin[i] = __float2bfloat16(v);
  }
  if (i < BB * HH) {
    const int b = i / HH, j = i % HH;
    const float v = h0[i];
    hF[i] = v;
    const __hip_bfloat16 hb = __float2bfloat16(v);
    hB[i] = hb;
    gin[b * 1024 + j] = hb;                 // gin = [h, x]
    repin[b * KREP + EMB + j] = hb;         // repin = [symlog(e), h]
  }
}

// ---------------- per-step prep: symlog(embed) + actions ----------------
__global__ void k_prep(const float* __restrict__ embed, const float* __restrict__ actions,
                       int t, __hip_bfloat16* __restrict__ repin,
                       __hip_bfloat16* __restrict__ zin)
{
  const int i = blockIdx.x * 256 + threadIdx.x;
  if (i < BB * EMB) {
    const int b = i >> 12, e = i & (EMB - 1);
    const float x = embed[((long)b * TT + t) * EMB + e];
    const float s = copysignf(log1pf(fabsf(x)), x);     // symlog
    repin[b * KREP + e] = __float2bfloat16(s);
    if (e < ACTD)
      zin[b * KREC + S2 + e] =
          __float2bfloat16(actions[((long)b * TT + t) * ACTD + e]);
  }
}

// ---------------- GRU gating: h' = u*tanh(r*c) + (1-u)*h ----------------
__global__ void k_gate(const float* __restrict__ gx, float* __restrict__ hF,
                       __hip_bfloat16* __restrict__ hB,
                       __hip_bfloat16* __restrict__ gin,
                       __hip_bfloat16* __restrict__ repin,
                       float* __restrict__ hsBase /* d_out hs + t*512 */)
{
  const int i = blockIdx.x * 256 + threadIdx.x;
  if (i < BB * HH) {
    const int b = i >> 9, j = i & (HH - 1);
    const float* row = gx + (long)b * G3;
    const float r = 1.0f / (1.0f + __expf(-row[j]));
    const float c = tanhf(r * row[HH + j]);
    const float u = 1.0f / (1.0f + __expf(-(row[2 * HH + j] - 1.0f)));
    const float hn = u * c + (1.0f - u) * hF[i];
    hF[i] = hn;
    hsBase[(long)b * (TT * HH) + j] = hn;
    const __hip_bfloat16 hb = __float2bfloat16(hn);
    hB[i] = hb;
    gin[b * 1024 + j] = hb;
    repin[b * KREP + EMB + j] = hb;
  }
}

// ---------------- unimix softmax + gumbel argmax + straight-through ----------
__global__ void k_sample(const float* __restrict__ postsBase,   // + t*1024
                         const float* __restrict__ gumBase,     // + t*1024
                         float* __restrict__ zsBase,            // + t*1024
                         __hip_bfloat16* __restrict__ zin)
{
  const int g = blockIdx.x * 256 + threadIdx.x;      // 256*32 rows * 32 lanes
  const int lane = g & 31;
  const int row  = g >> 5;
  if (row < BB * 32) {
    const int b = row >> 5, ir = row & 31;
    const long off = (long)b * (TT * S2) + ir * 32 + lane;
    const float p = postsBase[off];
    float m = p;
    for (int o = 16; o; o >>= 1) m = fmaxf(m, __shfl_xor(m, o, 32));
    const float e = __expf(p - m);
    float s = e;
    for (int o = 16; o; o >>= 1) s += __shfl_xor(s, o, 32);
    const float pr = 0.99f * (e / s) + 0.01f / 32.0f;
    const float lp = __logf(pr) + gumBase[off];
    float bv = lp; int bi = lane;
    for (int o = 16; o; o >>= 1) {
      const float ov = __shfl_xor(bv, o, 32);
      const int   oi = __shfl_xor(bi, o, 32);
      if (ov > bv || (ov == bv && oi < bi)) { bv = ov; bi = oi; }
    }
    const float one = (lane == bi) ? 1.0f : 0.0f;
    const float z = one + pr - pr;                   // straight-through fwd value
    zsBase[off] = z;
    zin[b * KREC + ir * 32 + lane] = __float2bfloat16(z);
  }
}

// ---------------- host ----------------
extern "C" void kernel_launch(void* const* d_in, const int* in_sizes, int n_in,
                              void* d_out, int out_size, void* d_ws, size_t ws_size,
                              hipStream_t stream)
{
  (void)in_sizes; (void)n_in; (void)out_size; (void)ws_size;
  auto F = [&](int i) { return (const float*)d_in[i]; };

  const float* embed   = F(0);
  const float* actions = F(1);
  const float* h0      = F(2);
  const float* z0      = F(3);
  const float* gumbel  = F(4);
  // params in setup_inputs insertion order:
  // rec l0..l2 (W,b,g,be)=5..16, rec out (W,b)=17,18, gru (W,b,g,be)=19..22,
  // rep l0..l2=23..34, rep out=35,36, trans l0..l2=37..48, trans out=49,50

  float* out   = (float*)d_out;
  float* hs    = out;                              // [256,64,512]
  float* posts = out + (long)BB * TT * HH;         // [256,64,1024]
  float* prio  = posts + (long)BB * TT * S2;
  float* zs    = prio + (long)BB * TT * S2;

  // workspace carve (deterministic)
  char* ws = (char*)d_ws;
  size_t woff = 0;
  auto carve = [&](size_t bytes) -> void* {
    void* p = ws + woff;
    woff = (woff + bytes + 255) & ~(size_t)255;
    return p;
  };
  typedef __hip_bfloat16 bf;
  bf* recW0 = (bf*)carve((size_t)512 * KREC * 2);
  bf* recW1 = (bf*)carve((size_t)512 * 512 * 2);
  bf* recW2 = (bf*)carve((size_t)512 * 512 * 2);
  bf* recWo = (bf*)carve((size_t)512 * 512 * 2);
  bf* gruW  = (bf*)carve((size_t)G3 * 1024 * 2);
  bf* repW0 = (bf*)carve((size_t)512 * KREP * 2);
  bf* repW1 = (bf*)carve((size_t)512 * 512 * 2);
  bf* repW2 = (bf*)carve((size_t)512 * 512 * 2);
  bf* repWo = (bf*)carve((size_t)1024 * 512 * 2);
  bf* trW0  = (bf*)carve((size_t)512 * 512 * 2);
  bf* trW1  = (bf*)carve((size_t)512 * 512 * 2);
  bf* trW2  = (bf*)carve((size_t)512 * 512 * 2);
  bf* trWo  = (bf*)carve((size_t)1024 * 512 * 2);
  bf* zin   = (bf*)carve((size_t)BB * KREC * 2);
  bf* gin   = (bf*)carve((size_t)BB * 1024 * 2);
  bf* repin = (bf*)carve((size_t)BB * KREP * 2);
  bf* hB    = (bf*)carve((size_t)BB * HH * 2);
  bf* act0  = (bf*)carve((size_t)BB * HH * 2);
  bf* act1  = (bf*)carve((size_t)BB * HH * 2);
  float* gx = (float*)carve((size_t)BB * G3 * 4);
  float* hF = (float*)carve((size_t)BB * HH * 4);

  auto cvt = [&](int widx, bf* dst, int K, int N, int KP) {
    const long n = (long)N * KP;
    k_wt<<<dim3((unsigned)((n + 255) / 256)), dim3(256), 0, stream>>>(F(widx), dst, K, N, KP);
  };
  cvt(5,  recW0, 1030, 512, KREC);
  cvt(9,  recW1, 512, 512, 512);
  cvt(13, recW2, 512, 512, 512);
  cvt(17, recWo, 512, 512, 512);
  cvt(19, gruW, 1024, G3, 1024);
  cvt(23, repW0, KREP, 512, KREP);
  cvt(27, repW1, 512, 512, 512);
  cvt(31, repW2, 512, 512, 512);
  cvt(35, repWo, 512, 1024, 512);
  cvt(37, trW0, 512, 512, 512);
  cvt(41, trW1, 512, 512, 512);
  cvt(45, trW2, 512, 512, 512);
  cvt(49, trWo, 512, 1024, 512);

  k_init<<<dim3((BB * KREC + 255) / 256), dim3(256), 0, stream>>>(h0, z0, hF, hB, gin, zin, repin);

  const dim3 gg(16), gb(256);
#define G_LNSILU(Ain, LDA, W_, bi, ga, be_, Oout)                                   \
  gemm16<EPI_LNSILU_BF16, 32><<<gg, gb, 0, stream>>>(Ain, LDA, W_, F(bi), F(ga),    \
                                                     F(be_), nullptr, Oout, 512, 0)

  for (int t = 0; t < TT; ++t) {
    k_prep<<<dim3((BB * EMB + 255) / 256), gb, 0, stream>>>(embed, actions, t, repin, zin);

    // ---- rec_mlp: [z,a] -> x ----
    G_LNSILU(zin, KREC, recW0, 6, 7, 8, act0);
    G_LNSILU(act0, 512, recW1, 10, 11, 12, act1);
    G_LNSILU(act1, 512, recW2, 14, 15, 16, act0);
    gemm16<EPI_BIAS_BF16, 32><<<gg, gb, 0, stream>>>(act0, 512, recWo, F(18),
        nullptr, nullptr, nullptr, gin, 1024, 512);           // x -> gin[:,512:]

    // ---- GRU: ln(gin @ W + b) then gates ----
    gemm16<EPI_LN_F32, 96><<<gg, gb, 0, stream>>>(gin, 1024, gruW, F(20), F(21),
        F(22), gx, nullptr, G3, 0);
    k_gate<<<dim3((BB * HH + 255) / 256), gb, 0, stream>>>(gx, hF, hB, gin, repin,
        hs + (long)t * HH);

    // ---- rep_mlp: [symlog(e), h] -> post ----
    G_LNSILU(repin, KREP, repW0, 24, 25, 26, act0);
    G_LNSILU(act0, 512, repW1, 28, 29, 30, act1);
    G_LNSILU(act1, 512, repW2, 32, 33, 34, act0);
    gemm16<EPI_BIAS_F32, 64><<<gg, gb, 0, stream>>>(act0, 512, repWo, F(36),
        nullptr, nullptr, posts + (long)t * S2, nullptr, TT * S2, 0);

    // ---- trans_mlp: h -> prior ----
    G_LNSILU(hB, 512, trW0, 38, 39, 40, act0);
    G_LNSILU(act0, 512, trW1, 42, 43, 44, act1);
    G_LNSILU(act1, 512, trW2, 46, 47, 48, act0);
    gemm16<EPI_BIAS_F32, 64><<<gg, gb, 0, stream>>>(act0, 512, trWo, F(50),
        nullptr, nullptr, prio + (long)t * S2, nullptr, TT * S2, 0);

    // ---- sample z_t ----
    k_sample<<<dim3((BB * 32 * 32 + 255) / 256), gb, 0, stream>>>(
        posts + (long)t * S2, gumbel + (long)t * S2, zs + (long)t * S2, zin);
  }
#undef G_LNSILU
}
